// ZigMoEStudent_71932112273437
// MI455X (gfx1250) — matrix operand order repaired and checked by reading, not compiled, not run on hardware
//
#include <hip/hip_runtime.h>

#define H_DIM 896
#define F_DIM 4864
#define E_NUM 8
#define S_DIM 8192

typedef __bf16 bf16_t;
typedef bf16_t v16bf __attribute__((ext_vector_type(16)));
typedef float v8f __attribute__((ext_vector_type(8)));

union Frag {
  v16bf v;
  uint4 q[2];
};

__device__ __forceinline__ unsigned short f32_to_bf16(float f) {
  unsigned int u = __float_as_uint(f);
  u += 0x7FFFu + ((u >> 16) & 1u);   // round-to-nearest-even
  return (unsigned short)(u >> 16);
}

__device__ __forceinline__ int expert_of(int f) {
  // cumulative EXPERT_SIZES boundaries: 560,1136,1728,2336,2944,3568,4208,4864
  return (f >= 560) + (f >= 1136) + (f >= 1728) + (f >= 2336) +
         (f >= 2944) + (f >= 3568) + (f >= 4208);
}

// ---------------------------------------------------------------------------
// LDS tile fragment loaders.  Tiles are [rows][32] bf16, row stride 40 elems
// (80 B = 5*16 B: every 16-B chunk is 16-B aligned and the 16 lanes of a
// b128 read land on disjoint bank groups).
//
// A-matrix 16x32 bf16 per-lane layout (CDNA5 ISA 7.12.2):
//   lane L(0-15):  M=L, elems 0..7 -> K=0..7,  elems 8..15 -> K=16..23
//   lane L+16:     M=L, elems 0..7 -> K=8..15, elems 8..15 -> K=24..31
__device__ __forceinline__ v16bf load_a_frag(const unsigned short* tile, int row, int lane) {
  Frag fr;
  const unsigned short* p = tile + row * 40 + ((lane & 16) ? 8 : 0);
  fr.q[0] = *(const uint4*)(p);
  fr.q[1] = *(const uint4*)(p + 16);
  return fr.v;
}

// B-matrix 32x16 bf16 per-lane layout:
//   lane L(0-15): N=L, elems 0..7 -> K=0..7,  elems 8..15 -> K=16..23
//   lane L+16:     N=L, elems 0..7 -> K=8..15, elems 8..15 -> K=24..31
__device__ __forceinline__ v16bf load_b_frag(const unsigned short* tile, int col, int lane) {
  Frag fr;
  const unsigned short* p = tile + col * 40 + ((lane & 16) ? 8 : 0);
  fr.q[0] = *(const uint4*)(p);
  fr.q[1] = *(const uint4*)(p + 16);
  return fr.v;
}

// ---------------------------------------------------------------------------
// f32 -> bf16 conversion, 4 elements / thread (all sizes divisible by 1024)
__global__ void __launch_bounds__(256) cvt_kernel(const float* __restrict__ src,
                                                  unsigned short* __restrict__ dst) {
  int i = (blockIdx.x * 256 + threadIdx.x) * 4;
  float4 v = *(const float4*)(src + i);
  ushort4 o;
  o.x = f32_to_bf16(v.x);
  o.y = f32_to_bf16(v.y);
  o.z = f32_to_bf16(v.z);
  o.w = f32_to_bf16(v.w);
  *(ushort4*)(dst + i) = o;
}

// ---------------------------------------------------------------------------
// Router: one wave32 per token; softmax over 8 experts.
__global__ void __launch_bounds__(256) router_kernel(const float* __restrict__ x,
                                                     const float* __restrict__ rw,
                                                     float* __restrict__ probs) {
  const int wave = threadIdx.x >> 5;
  const int lane = threadIdx.x & 31;
  const int token = blockIdx.x * 8 + wave;
  const float* xr = x + (size_t)token * H_DIM;

  float acc[E_NUM];
#pragma unroll
  for (int e = 0; e < E_NUM; ++e) acc[e] = 0.0f;

  for (int h = lane; h < H_DIM; h += 32) {
    float xv = xr[h];
#pragma unroll
    for (int e = 0; e < E_NUM; ++e) acc[e] += xv * rw[e * H_DIM + h];
  }
#pragma unroll
  for (int e = 0; e < E_NUM; ++e) {
#pragma unroll
    for (int off = 16; off > 0; off >>= 1) acc[e] += __shfl_xor(acc[e], off, 32);
  }
  if (lane == 0) {
    float m = acc[0];
#pragma unroll
    for (int e = 1; e < E_NUM; ++e) m = fmaxf(m, acc[e]);
    float ex[E_NUM];
    float ssum = 0.0f;
#pragma unroll
    for (int e = 0; e < E_NUM; ++e) { ex[e] = __expf(acc[e] - m); ssum += ex[e]; }
    float inv = __builtin_amdgcn_rcpf(ssum);
#pragma unroll
    for (int e = 0; e < E_NUM; ++e) probs[token * E_NUM + e] = ex[e] * inv;
  }
}

// ---------------------------------------------------------------------------
// MLP1: inter[s,f] = silu(x@wg^T) * (x@wu^T) * probs[s, expert(f)]  (bf16 out)
// Block tile 128(M=s) x 128(N=f), 8 waves as 2(M) x 4(N); each wave 64x32
// with gate+up accumulators -> 16 WMMAs / 8 fragment reads per K=32 step.
// Double-buffered LDS, one __syncthreads per K step.
__global__ void __launch_bounds__(256) mlp1_kernel(const unsigned short* __restrict__ xb,
                                                   const unsigned short* __restrict__ wgb,
                                                   const unsigned short* __restrict__ wub,
                                                   const float* __restrict__ probs,
                                                   unsigned short* __restrict__ inter) {
  // [buf][ A:128x40 | Bgate:128x40 | Bup:128x40 ]
  __shared__ __attribute__((aligned(16))) unsigned short lds[2][15360];

  const int f0 = blockIdx.x * 128;
  const int s0 = blockIdx.y * 128;
  const int t = threadIdx.x;
  const int lane = t & 31;
  const int w = t >> 5;
  const int wm = w & 1;   // M slice (64 rows)
  const int wn = w >> 1;  // N slice (32 cols)

  v8f accg[4][2] = {};
  v8f accu[4][2] = {};

  // cooperative copy plan: each thread moves one 32-B chunk of A, Bgate, Bup
  const int row = t >> 1;            // 0..127
  const int chk = (t & 1) * 16;      // element offset within 32-elem row
  const unsigned short* asrc = xb + (size_t)(s0 + row) * H_DIM + chk;
  const unsigned short* gsrc = wgb + (size_t)(f0 + row) * H_DIM + chk;
  const unsigned short* usrc = wub + (size_t)(f0 + row) * H_DIM + chk;
  const int ldst = row * 40 + chk;

  // prologue: tile 0 -> buffer 0
  {
    const uint4* ga = (const uint4*)(asrc);
    const uint4* gg = (const uint4*)(gsrc);
    const uint4* gu = (const uint4*)(usrc);
    uint4* da = (uint4*)&lds[0][ldst];
    uint4* dg = (uint4*)&lds[0][5120 + ldst];
    uint4* du = (uint4*)&lds[0][10240 + ldst];
    da[0] = ga[0]; da[1] = ga[1];
    dg[0] = gg[0]; dg[1] = gg[1];
    du[0] = gu[0]; du[1] = gu[1];
  }

  const int KT = H_DIM / 32;  // 28
  for (int kt = 0; kt < KT; ++kt) {
    const int cur = kt & 1;
    uint4 ra0, ra1, rg0, rg1, ru0, ru1;
    const bool more = (kt + 1 < KT);
    if (more) {
      const int k0 = (kt + 1) * 32;
      const uint4* ga = (const uint4*)(asrc + k0);
      const uint4* gg = (const uint4*)(gsrc + k0);
      const uint4* gu = (const uint4*)(usrc + k0);
      ra0 = ga[0]; ra1 = ga[1];
      rg0 = gg[0]; rg1 = gg[1];
      ru0 = gu[0]; ru1 = gu[1];
      __builtin_prefetch(asrc + k0 + 32, 0, 1);
      __builtin_prefetch(gsrc + k0 + 32, 0, 1);
      __builtin_prefetch(usrc + k0 + 32, 0, 1);
    }
    __syncthreads();

    const unsigned short* L = lds[cur];
    v16bf af[4], bg[2], bu[2];
#pragma unroll
    for (int mi = 0; mi < 4; ++mi)
      af[mi] = load_a_frag(L, 64 * wm + 16 * mi + (lane & 15), lane);
#pragma unroll
    for (int ni = 0; ni < 2; ++ni) {
      bg[ni] = load_b_frag(L + 5120,  32 * wn + 16 * ni + (lane & 15), lane);
      bu[ni] = load_b_frag(L + 10240, 32 * wn + 16 * ni + (lane & 15), lane);
    }
#pragma unroll
    for (int mi = 0; mi < 4; ++mi)
#pragma unroll
      for (int ni = 0; ni < 2; ++ni) {
        accg[mi][ni] = __builtin_amdgcn_wmma_f32_16x16x32_bf16(
            false, af[mi], false, bg[ni], (short)0, accg[mi][ni], false, false);
        accu[mi][ni] = __builtin_amdgcn_wmma_f32_16x16x32_bf16(
            false, af[mi], false, bu[ni], (short)0, accu[mi][ni], false, false);
      }

    if (more) {
      const int nxt = cur ^ 1;
      uint4* da = (uint4*)&lds[nxt][ldst];
      uint4* dg = (uint4*)&lds[nxt][5120 + ldst];
      uint4* du = (uint4*)&lds[nxt][10240 + ldst];
      da[0] = ra0; da[1] = ra1;
      dg[0] = rg0; dg[1] = rg1;
      du[0] = ru0; du[1] = ru1;
    }
  }

  // epilogue: silu(gate) * up * prob -> bf16
#pragma unroll
  for (int mi = 0; mi < 4; ++mi) {
#pragma unroll
    for (int ni = 0; ni < 2; ++ni) {
      const int f = f0 + 32 * wn + 16 * ni + (lane & 15);
      const int eid = expert_of(f);
#pragma unroll
      for (int r = 0; r < 8; ++r) {
        const int s = s0 + 64 * wm + 16 * mi + r + ((lane & 16) ? 8 : 0);
        float g = accg[mi][ni][r];
        float u = accu[mi][ni][r];
        float p = probs[s * E_NUM + eid];
        float sig = __builtin_amdgcn_rcpf(1.0f + __expf(-g));
        inter[(size_t)s * F_DIM + f] = f32_to_bf16(g * sig * u * p);
      }
    }
  }
}

// ---------------------------------------------------------------------------
// MLP2: out[s,h] = inter @ w_down^T.  Block tile 128(M=s) x 128(N=h),
// 8 waves as 2(M) x 4(N); each wave 64x32 -> 8 WMMAs per K=32 step.
__global__ void __launch_bounds__(256) mlp2_kernel(const unsigned short* __restrict__ interb,
                                                   const unsigned short* __restrict__ wdb,
                                                   float* __restrict__ out) {
  // [buf][ A:128x40 | B:128x40 ]
  __shared__ __attribute__((aligned(16))) unsigned short lds[2][10240];

  const int h0 = blockIdx.x * 128;
  const int s0 = blockIdx.y * 128;
  const int t = threadIdx.x;
  const int lane = t & 31;
  const int w = t >> 5;
  const int wm = w & 1;   // M slice (64 rows)
  const int wn = w >> 1;  // N slice (32 cols)

  v8f acc[4][2] = {};

  const int row = t >> 1;
  const int chk = (t & 1) * 16;
  const unsigned short* asrc = interb + (size_t)(s0 + row) * F_DIM + chk;
  const unsigned short* bsrc = wdb + (size_t)(h0 + row) * F_DIM + chk;
  const int ldst = row * 40 + chk;

  {
    const uint4* ga = (const uint4*)(asrc);
    const uint4* gb = (const uint4*)(bsrc);
    uint4* da = (uint4*)&lds[0][ldst];
    uint4* db = (uint4*)&lds[0][5120 + ldst];
    da[0] = ga[0]; da[1] = ga[1];
    db[0] = gb[0]; db[1] = gb[1];
  }

  const int KT = F_DIM / 32;  // 152
  for (int kt = 0; kt < KT; ++kt) {
    const int cur = kt & 1;
    uint4 ra0, ra1, rb0, rb1;
    const bool more = (kt + 1 < KT);
    if (more) {
      const int k0 = (kt + 1) * 32;
      const uint4* ga = (const uint4*)(asrc + k0);
      const uint4* gb = (const uint4*)(bsrc + k0);
      ra0 = ga[0]; ra1 = ga[1];
      rb0 = gb[0]; rb1 = gb[1];
      __builtin_prefetch(asrc + k0 + 32, 0, 1);
      __builtin_prefetch(bsrc + k0 + 32, 0, 1);
    }
    __syncthreads();

    const unsigned short* L = lds[cur];
    v16bf af[4], bf[2];
#pragma unroll
    for (int mi = 0; mi < 4; ++mi)
      af[mi] = load_a_frag(L, 64 * wm + 16 * mi + (lane & 15), lane);
#pragma unroll
    for (int ni = 0; ni < 2; ++ni)
      bf[ni] = load_b_frag(L + 5120, 32 * wn + 16 * ni + (lane & 15), lane);
#pragma unroll
    for (int mi = 0; mi < 4; ++mi)
#pragma unroll
      for (int ni = 0; ni < 2; ++ni)
        acc[mi][ni] = __builtin_amdgcn_wmma_f32_16x16x32_bf16(
            false, af[mi], false, bf[ni], (short)0, acc[mi][ni], false, false);

    if (more) {
      const int nxt = cur ^ 1;
      uint4* da = (uint4*)&lds[nxt][ldst];
      uint4* db = (uint4*)&lds[nxt][5120 + ldst];
      da[0] = ra0; da[1] = ra1;
      db[0] = rb0; db[1] = rb1;
    }
  }

#pragma unroll
  for (int mi = 0; mi < 4; ++mi) {
#pragma unroll
    for (int ni = 0; ni < 2; ++ni) {
      const int h = h0 + 32 * wn + 16 * ni + (lane & 15);
#pragma unroll
      for (int r = 0; r < 8; ++r) {
        const int s = s0 + 64 * wm + 16 * mi + r + ((lane & 16) ? 8 : 0);
        out[(size_t)s * H_DIM + h] = acc[mi][ni][r];
      }
    }
  }
}

// ---------------------------------------------------------------------------
extern "C" void kernel_launch(void* const* d_in, const int* in_sizes, int n_in,
                              void* d_out, int out_size, void* d_ws, size_t ws_size,
                              hipStream_t stream) {
  (void)in_sizes; (void)n_in; (void)out_size; (void)ws_size;
  const float* x  = (const float*)d_in[0];
  const float* rw = (const float*)d_in[1];
  const float* wg = (const float*)d_in[2];
  const float* wu = (const float*)d_in[3];
  const float* wd = (const float*)d_in[4];
  float* out = (float*)d_out;

  char* base = (char*)d_ws;
  size_t off = 0;
  auto take = [&](size_t bytes) -> char* {
    char* p = base + off;
    off += (bytes + 255) & ~(size_t)255;
    return p;
  };
  float* probs           = (float*)take((size_t)S_DIM * E_NUM * sizeof(float));
  unsigned short* xb     = (unsigned short*)take((size_t)S_DIM * H_DIM * 2);
  unsigned short* wgb    = (unsigned short*)take((size_t)F_DIM * H_DIM * 2);
  unsigned short* wub    = (unsigned short*)take((size_t)F_DIM * H_DIM * 2);
  unsigned short* wdb    = (unsigned short*)take((size_t)H_DIM * F_DIM * 2);
  unsigned short* interb = (unsigned short*)take((size_t)S_DIM * F_DIM * 2);

  const int nx = S_DIM * H_DIM;   // 7,340,032  (div by 1024)
  const int nw = F_DIM * H_DIM;   // 4,358,144  (div by 1024)
  cvt_kernel<<<nx / 1024, 256, 0, stream>>>(x, xb);
  cvt_kernel<<<nw / 1024, 256, 0, stream>>>(wg, wgb);
  cvt_kernel<<<nw / 1024, 256, 0, stream>>>(wu, wub);
  cvt_kernel<<<nw / 1024, 256, 0, stream>>>(wd, wdb);
  router_kernel<<<S_DIM / 8, 256, 0, stream>>>(x, rw, probs);
  mlp1_kernel<<<dim3(F_DIM / 128, S_DIM / 128), 256, 0, stream>>>(xb, wgb, wub, probs, interb);
  mlp2_kernel<<<dim3(H_DIM / 128, S_DIM / 128), 256, 0, stream>>>(interb, wdb, out);
}
